// OFAMLP_75050258530468
// MI455X (gfx1250) — matrix operand order repaired and checked by reading, not compiled, not run on hardware
//
#include <hip/hip_runtime.h>
#include <hip/hip_bf16.h>
#include <stdint.h>
#include <stddef.h>

// ---------------------------------------------------------------------------
// OFA-MLP for MI455X (gfx1250): f16 WMMA w/ f32 accumulate.
// Only the 144 live rows per graph are projected; edge_attr branch is dead.
// ---------------------------------------------------------------------------

typedef __attribute__((ext_vector_type(16))) _Float16 v16h;
typedef __attribute__((ext_vector_type(8)))  float    v8f;
typedef __attribute__((ext_vector_type(8)))  uint32_t v8u;

union Frag16 { v16h h; v8u u; };

static constexpr int G      = 512;   // graphs
static constexpr int P      = 512;   // nodes per graph
static constexpr int INDIM  = 512;
static constexpr int OUTDIM = 256;
static constexpr int NCLS   = 16;    // class ("true") nodes per graph
static constexpr int FEATD  = 512;   // 2*OUTDIM
static constexpr int MROWS  = G * NCLS;  // 8192 MLP rows
static constexpr int TASK   = 40;
static constexpr int TASKP  = 48;    // padded to 3 n-tiles

// ---- WMMA wrapper --------------------------------------------------------
__device__ __forceinline__ v8f wmma_f16(v16h a, v16h b, v8f c) {
  return __builtin_amdgcn_wmma_f32_16x16x32_f16(
      /*neg_a=*/false, a, /*neg_b=*/false, b,
      /*c_mod=*/(short)0, c, /*reuse_a=*/false, /*reuse_b=*/false);
}

// ---- A fragment (16x32 f16) from an fp32 row-major tile (src at [m0,k0]) --
__device__ __forceinline__ v16h load_a_f32(const float* __restrict__ src,
                                           int ld, int lane) {
  const int m = lane & 15, half = lane >> 4;
  const float* row = src + (size_t)m * ld;
  v16h a;
#pragma unroll
  for (int v = 0; v < 8; ++v) {
    const int kk = (v < 4) ? (half * 8 + 2 * v) : (16 + half * 8 + 2 * (v - 4));
    const float2 f = *(const float2*)(row + kk);
    a[2 * v]     = (_Float16)f.x;
    a[2 * v + 1] = (_Float16)f.y;
  }
  return a;
}

// ---- A fragment from an f16 row-major tile -------------------------------
__device__ __forceinline__ v16h load_a_f16(const _Float16* __restrict__ src,
                                           int ld, int lane) {
  const int m = lane & 15, half = lane >> 4;
  const _Float16* row = src + (size_t)m * ld;
  Frag16 a;
#pragma unroll
  for (int v = 0; v < 8; ++v) {
    const int kk = (v < 4) ? (half * 8 + 2 * v) : (16 + half * 8 + 2 * (v - 4));
    a.u[v] = *(const uint32_t*)(row + kk);
  }
  return a.h;
}

// ---- B fragment (32x16 f16) from TRANSPOSED weights Wt[N][K], at (k0,n0) --
// wt must already point at Wt + n0*ldk + k0.
__device__ __forceinline__ v16h load_b_t(const _Float16* __restrict__ wt,
                                         int ldk, int lane) {
  const int n = lane & 15, half = lane >> 4;
  const _Float16* col = wt + (size_t)n * ldk;
  Frag16 b;
#pragma unroll
  for (int v = 0; v < 8; ++v) {
    const int kk = (v < 4) ? (half * 8 + 2 * v) : (16 + half * 8 + 2 * (v - 4));
    b.u[v] = *(const uint32_t*)(col + kk);
  }
  return b.h;
}

// ---------------------------------------------------------------------------
// Weight convert + transpose: W[K][N] fp32 -> Wt[Npad][K] f16 (zero padding).
// ---------------------------------------------------------------------------
__global__ void cvt_transpose_kernel(const float* __restrict__ W,
                                     _Float16* __restrict__ Wt,
                                     int K, int N, int Npad) {
  const int idx = blockIdx.x * blockDim.x + threadIdx.x;
  if (idx >= Npad * K) return;
  const int n = idx / K;
  const int k = idx - n * K;
  const float v = (n < N) ? W[(size_t)k * N + n] : 0.0f;
  Wt[idx] = (_Float16)v;
}

// ---------------------------------------------------------------------------
// Projection: per graph, H = x[g, 0:144, :] @ W_in (+b_in).
//   m-tile 0  (rows 0..15, true nodes)   -> feat[:, 256:512]
//   m-tiles 1..8 (rows 16..143, targets) -> column mean -> feat[:, 0:256]
// Block = 256 threads (8 waves); wave w owns n-tiles {2w, 2w+1}.
// ---------------------------------------------------------------------------
__global__ void proj_kernel(const float* __restrict__ x,
                            const _Float16* __restrict__ WtIn,  // [256][512]
                            const float* __restrict__ b_in,
                            _Float16* __restrict__ feat) {      // [8192][512]
  const int g    = blockIdx.x;
  const int wave = threadIdx.x >> 5;
  const int lane = threadIdx.x & 31;
  const int n0   = wave * 32;  // two 16-col tiles: n0, n0+16
  const float* xg = x + (size_t)g * P * INDIM;

  float s0 = 0.0f, s1 = 0.0f;  // target-row sum partials (per-lane)

#pragma unroll 1
  for (int mt = 0; mt < 9; ++mt) {
    v8f acc0 = {}; v8f acc1 = {};
    const float* asrc = xg + (size_t)(mt * 16) * INDIM;
#pragma unroll 2
    for (int kt = 0; kt < INDIM / 32; ++kt) {
      const v16h a  = load_a_f32(asrc + kt * 32, INDIM, lane);
      const v16h b0 = load_b_t(WtIn + (size_t)n0 * INDIM + kt * 32, INDIM, lane);
      const v16h b1 = load_b_t(WtIn + (size_t)(n0 + 16) * INDIM + kt * 32, INDIM, lane);
      acc0 = wmma_f16(a, b0, acc0);
      acc1 = wmma_f16(a, b1, acc1);
    }
    const int ln = lane & 15, half = lane >> 4;
    if (mt == 0) {
      // true-node rows -> right half of feat
      const int c0 = n0 + ln, c1 = n0 + 16 + ln;
      const float bb0 = b_in[c0], bb1 = b_in[c1];
#pragma unroll
      for (int v = 0; v < 8; ++v) {
        const int row = g * NCLS + v + 8 * half;
        feat[(size_t)row * FEATD + OUTDIM + c0] = (_Float16)(acc0[v] + bb0);
        feat[(size_t)row * FEATD + OUTDIM + c1] = (_Float16)(acc1[v] + bb1);
      }
    } else {
#pragma unroll
      for (int v = 0; v < 8; ++v) { s0 += acc0[v]; s1 += acc1[v]; }
    }
  }

  // combine the two half-lane groups (rows 0..7 vs 8..15 of each tile)
  s0 += __shfl_xor(s0, 16, 32);
  s1 += __shfl_xor(s1, 16, 32);
  if (lane < 16) {
    const int c0 = n0 + lane, c1 = n0 + 16 + lane;
    const _Float16 e0 = (_Float16)(s0 * (1.0f / 128.0f) + b_in[c0]);
    const _Float16 e1 = (_Float16)(s1 * (1.0f / 128.0f) + b_in[c1]);
#pragma unroll
    for (int c = 0; c < NCLS; ++c) {
      feat[(size_t)(g * NCLS + c) * FEATD + c0] = e0;
      feat[(size_t)(g * NCLS + c) * FEATD + c1] = e1;
    }
  }
}

// ---------------------------------------------------------------------------
// MLP layer: Out[M][N] = act(A[M][K] @ Wt^T + bias), f16 activations.
// Block = 256 threads (8 waves); wave w owns NT consecutive n-tiles.
// grid.x = M/16 m-tiles.
// ---------------------------------------------------------------------------
template <int NT, bool RELU>
__global__ void mlp_kernel(const _Float16* __restrict__ A,   // [M][K]
                           const _Float16* __restrict__ Wt,  // [N][K]
                           const float* __restrict__ bias,
                           _Float16* __restrict__ Out,       // [M][N]
                           int K, int N) {
  const int mt   = blockIdx.x;
  const int wave = threadIdx.x >> 5;
  const int lane = threadIdx.x & 31;
  const int n0   = wave * (NT * 16);

  v8f acc[NT];
#pragma unroll
  for (int t = 0; t < NT; ++t) acc[t] = (v8f){};

  const _Float16* arow = A + (size_t)mt * 16 * K;
  const int ksteps = K >> 5;
#pragma unroll 2
  for (int kt = 0; kt < ksteps; ++kt) {
    const v16h a = load_a_f16(arow + kt * 32, K, lane);
#pragma unroll
    for (int t = 0; t < NT; ++t) {
      const v16h b = load_b_t(Wt + (size_t)(n0 + t * 16) * K + kt * 32, K, lane);
      acc[t] = wmma_f16(a, b, acc[t]);
    }
  }

  const int ln = lane & 15, half = lane >> 4;
#pragma unroll
  for (int t = 0; t < NT; ++t) {
    const int col = n0 + t * 16 + ln;
    const float bb = bias[col];
#pragma unroll
    for (int v = 0; v < 8; ++v) {
      const int row = mt * 16 + v + 8 * half;
      float val = acc[t][v] + bb;
      if (RELU) val = fmaxf(val, 0.0f);
      Out[(size_t)row * N + col] = (_Float16)val;
    }
  }
}

// ---------------------------------------------------------------------------
// Head: out[8192][40] = z2[8192][256] @ W3 + b3 (fp32 store, N padded to 48).
// Block = 96 threads (3 waves, one n-tile each). grid.x = 512 m-tiles.
// ---------------------------------------------------------------------------
__global__ void head_kernel(const _Float16* __restrict__ A,    // [8192][256]
                            const _Float16* __restrict__ Wt3,  // [48][256]
                            const float* __restrict__ b3,
                            float* __restrict__ out) {         // [8192][40]
  const int mt   = blockIdx.x;
  const int wave = threadIdx.x >> 5;
  const int lane = threadIdx.x & 31;
  const int n0   = wave * 16;

  v8f acc = {};
  const _Float16* arow = A + (size_t)mt * 16 * OUTDIM;
#pragma unroll
  for (int kt = 0; kt < OUTDIM / 32; ++kt) {
    const v16h a = load_a_f16(arow + kt * 32, OUTDIM, lane);
    const v16h b = load_b_t(Wt3 + (size_t)n0 * OUTDIM + kt * 32, OUTDIM, lane);
    acc = wmma_f16(a, b, acc);
  }

  const int ln = lane & 15, half = lane >> 4;
  const int col = n0 + ln;
  if (col < TASK) {
    const float bb = b3[col];
#pragma unroll
    for (int v = 0; v < 8; ++v) {
      const int row = mt * 16 + v + 8 * half;
      out[(size_t)row * TASK + col] = acc[v] + bb;
    }
  }
}

// ---------------------------------------------------------------------------
// Host launcher.
// Input order: x(0) edge_attr(1) batch(2) target_mask(3) true_mask(4)
//              num_graphs(5) num_classes(6) W_in(7) b_in(8) W1(9) b1(10)
//              W2(11) b2(12) W3(13) b3(14)
// ---------------------------------------------------------------------------
extern "C" void kernel_launch(void* const* d_in, const int* in_sizes, int n_in,
                              void* d_out, int out_size, void* d_ws, size_t ws_size,
                              hipStream_t stream) {
  (void)in_sizes; (void)n_in; (void)out_size; (void)ws_size;

  const float* x    = (const float*)d_in[0];
  const float* W_in = (const float*)d_in[7];
  const float* b_in = (const float*)d_in[8];
  const float* W1   = (const float*)d_in[9];
  const float* b1   = (const float*)d_in[10];
  const float* W2   = (const float*)d_in[11];
  const float* b2   = (const float*)d_in[12];
  const float* W3   = (const float*)d_in[13];
  const float* b3   = (const float*)d_in[14];
  float* out = (float*)d_out;

  // workspace carve-up (f16 elements)
  _Float16* WtIn = (_Float16*)d_ws;                    // [256][512]
  _Float16* Wt1  = WtIn + (size_t)OUTDIM * INDIM;      // [512][512]
  _Float16* Wt2  = Wt1  + (size_t)FEATD * FEATD;       // [256][512]
  _Float16* Wt3  = Wt2  + (size_t)OUTDIM * FEATD;      // [48][256]
  _Float16* feat = Wt3  + (size_t)TASKP * OUTDIM;      // [8192][512]
  _Float16* z1   = feat + (size_t)MROWS * FEATD;       // [8192][512]
  _Float16* z2   = z1   + (size_t)MROWS * FEATD;       // [8192][256]

  // 1) weights -> f16, transposed [N][K]
  {
    int e;
    e = OUTDIM * INDIM;  // W_in^T
    cvt_transpose_kernel<<<(e + 255) / 256, 256, 0, stream>>>(W_in, WtIn, INDIM, OUTDIM, OUTDIM);
    e = FEATD * FEATD;   // W1^T
    cvt_transpose_kernel<<<(e + 255) / 256, 256, 0, stream>>>(W1, Wt1, FEATD, FEATD, FEATD);
    e = OUTDIM * FEATD;  // W2^T
    cvt_transpose_kernel<<<(e + 255) / 256, 256, 0, stream>>>(W2, Wt2, FEATD, OUTDIM, OUTDIM);
    e = TASKP * OUTDIM;  // W3^T (padded 40 -> 48)
    cvt_transpose_kernel<<<(e + 255) / 256, 256, 0, stream>>>(W3, Wt3, OUTDIM, TASK, TASKP);
  }

  // 2) projection of the 144 live rows per graph -> feat [8192][512]
  proj_kernel<<<G, 256, 0, stream>>>(x, WtIn, b_in, feat);

  // 3) MLP
  mlp_kernel<4, true><<<MROWS / 16, 256, 0, stream>>>(feat, Wt1, b1, z1, FEATD, FEATD);
  mlp_kernel<2, true><<<MROWS / 16, 256, 0, stream>>>(z1, Wt2, b2, z2, FEATD, OUTDIM);

  // 4) head -> fp32 output [8192][40]
  head_kernel<<<MROWS / 16, 96, 0, stream>>>(z2, Wt3, b3, out);
}
